// GATLayer_68719476736403
// MI455X (gfx1250) — matrix operand order repaired and checked by reading, not compiled, not run on hardware
//
#include <hip/hip_runtime.h>
#include <hip/hip_bf16.h>

typedef __attribute__((ext_vector_type(16))) __bf16 v16bf;
typedef __attribute__((ext_vector_type(8)))  float  v8f;

#define NN 100000
#define NE 1600000
#define INN 128
#define INE 64
#define HD 64          // H*D
#define NTILES ((NN + 15) / 16)   // 6250
#define ETILES (NE / 16)          // 100000

// ---------- workspace layout (bytes) ----------
#define OFF_FRAG_N 0u               // 16 frags * 1KB = 16384
#define OFF_FRAG_E 16384u           // 8 frags * 1KB  = 8192
#define OFF_Z      32768u                       // N*64*4 = 25,600,000
#define OFF_EL     (OFF_Z + 25600000u)          // N*2*4 = 800,000
#define OFF_ER     (OFF_EL + 800000u)
#define OFF_EE     (OFF_ER + 800000u)           // E*2*4 = 12,800,000
#define OFF_LOGIT  (OFF_EE + 12800000u)         // E*2*4
#define OFF_M      (OFF_LOGIT + 12800000u)      // N*2*4
#define OFF_DENOM  (OFF_M + 800000u)

__device__ __forceinline__ unsigned enc_f(float f) {
    unsigned u = __float_as_uint(f);
    return (u & 0x80000000u) ? ~u : (u | 0x80000000u);
}
__device__ __forceinline__ float dec_f(unsigned u) {
    unsigned b = (u & 0x80000000u) ? (u ^ 0x80000000u) : ~u;
    return __uint_as_float(b);
}
__device__ __forceinline__ unsigned bf16bits(float f) {
    __bf16 h = (__bf16)f;
    return (unsigned)__builtin_bit_cast(unsigned short, h);
}

// ---------------------------------------------------------------------------
// Pack W_node [64,128] and W_edge [64,64] into WMMA-B bf16 fragments.
// Fragment (ot,kt): B[k, o], k in [kt*32,kt*32+32), o in [ot*16,ot*16+16).
// Lane l, dword v, half h: o = ot*16 + (l&15); k = kt*32 + ((l>>4)<<3) + ((v>>2)<<4) + 2(v&3) + h.
// Stored at frag[((ot*KT + kt)*32 + lane)*8 + v].
// ---------------------------------------------------------------------------
__global__ __launch_bounds__(256) void gat_pack_w(
    const float* __restrict__ Wn, const float* __restrict__ We,
    unsigned* __restrict__ fragN, unsigned* __restrict__ fragE)
{
    for (int idx = threadIdx.x; idx < 4096 + 2048; idx += 256) {
        bool isn = idx < 4096;
        int t    = isn ? idx : idx - 4096;
        int frag = t >> 8;          // 256 dwords per fragment
        int rem  = t & 255;
        int lane = rem >> 3;
        int v    = rem & 7;
        int KT   = isn ? 4 : 2;
        int K    = isn ? 128 : 64;
        int ot   = frag / KT;
        int kt   = frag % KT;
        const float* W = isn ? Wn : We;
        int col  = ot * 16 + (lane & 15);
        int k0   = kt * 32 + ((lane >> 4) << 3) + ((v >> 2) << 4) + 2 * (v & 3);
        unsigned lo = bf16bits(W[col * K + k0]);
        unsigned hi = bf16bits(W[col * K + k0 + 1]);
        unsigned* d = isn ? fragN : fragE;
        d[(size_t)(frag * 32 + lane) * 8 + v] = lo | (hi << 16);
    }
}

__global__ __launch_bounds__(256) void gat_init(
    float* __restrict__ n_out, const float* __restrict__ bias,
    unsigned* __restrict__ mten, float* __restrict__ denom)
{
    long i = (long)blockIdx.x * 256 + threadIdx.x;
    if (i < (long)NN * HD) n_out[i] = bias[i & 63];
    if (i < (long)NN * 2) { mten[i] = enc_f(-__builtin_inff()); denom[i] = 0.0f; }
}

// ---------------------------------------------------------------------------
// Node GEMM: z = nf @ W_node^T  (one wave per 16-row tile), + el/er reduction.
// ---------------------------------------------------------------------------
__global__ __launch_bounds__(256) void gat_node_gemm(
    const float* __restrict__ nf, const unsigned* __restrict__ frag,
    const float* __restrict__ attn_l, const float* __restrict__ attn_r,
    float* __restrict__ z, float* __restrict__ el, float* __restrict__ er)
{
    __shared__ float lds[8][16 * HD];
    int lane = threadIdx.x & 31;
    int wave = threadIdx.x >> 5;
    long tile = (long)blockIdx.x * 8 + wave;
    int r0 = (int)(tile * 16);
    bool active = r0 < NN;

    v8f acc[4] = {};
    if (active) {
        int row   = r0 + (lane & 15);
        int kbase = (lane < 16) ? 0 : 8;
        const float* arow = nf + (long)row * INN;
        #pragma unroll
        for (int kt = 0; kt < 4; ++kt) {
            v16bf a;
            const float* p0 = arow + kt * 32 + kbase;
            const float* p1 = p0 + 16;
            #pragma unroll
            for (int i = 0; i < 8; ++i) {
                a[i]     = (__bf16)p0[i];
                a[8 + i] = (__bf16)p1[i];
            }
            #pragma unroll
            for (int ot = 0; ot < 4; ++ot) {
                const v16bf b = *(const v16bf*)(frag + (size_t)((ot * 4 + kt) * 32 + lane) * 8);
                acc[ot] = __builtin_amdgcn_wmma_f32_16x16x32_bf16(
                    false, a, false, b, (short)0, acc[ot], false, false);
            }
        }
        int nn_ = lane & 15;
        int mb  = (lane < 16) ? 0 : 8;
        #pragma unroll
        for (int ot = 0; ot < 4; ++ot)
            #pragma unroll
            for (int r = 0; r < 8; ++r) {
                int m = mb + r, col = ot * 16 + nn_;
                float v = acc[ot][r];
                z[(long)(r0 + m) * HD + col] = v;
                lds[wave][m * HD + col] = v;
            }
    }
    __syncthreads();
    if (active) {
        int row = lane & 15, head = lane >> 4;
        const float* zr = &lds[wave][row * HD + head * 32];
        float sl = 0.f, sr = 0.f;
        #pragma unroll
        for (int d = 0; d < 32; ++d) {
            float v = zr[d];
            sl += v * attn_l[head * 32 + d];
            sr += v * attn_r[head * 32 + d];
        }
        el[(long)(r0 + row) * 2 + head] = sl;
        er[(long)(r0 + row) * 2 + head] = sr;
    }
}

// ---------------------------------------------------------------------------
// Edge GEMM: z_e = ef @ W_edge^T, e_out = z_e + bias, + ee reduction.
// ---------------------------------------------------------------------------
__global__ __launch_bounds__(256) void gat_edge_gemm(
    const float* __restrict__ ef, const unsigned* __restrict__ frag,
    const float* __restrict__ attn_e, const float* __restrict__ bias,
    float* __restrict__ e_out, float* __restrict__ ee)
{
    __shared__ float lds[8][16 * HD];
    int lane = threadIdx.x & 31;
    int wave = threadIdx.x >> 5;
    long tile = (long)blockIdx.x * 8 + wave;
    long r0 = tile * 16;
    bool active = r0 < (long)NE;

    v8f acc[4] = {};
    if (active) {
        long row  = r0 + (lane & 15);
        int kbase = (lane < 16) ? 0 : 8;
        const float* arow = ef + row * INE;
        #pragma unroll
        for (int kt = 0; kt < 2; ++kt) {
            v16bf a;
            const float* p0 = arow + kt * 32 + kbase;
            const float* p1 = p0 + 16;
            #pragma unroll
            for (int i = 0; i < 8; ++i) {
                a[i]     = (__bf16)p0[i];
                a[8 + i] = (__bf16)p1[i];
            }
            #pragma unroll
            for (int ot = 0; ot < 4; ++ot) {
                const v16bf b = *(const v16bf*)(frag + (size_t)((ot * 2 + kt) * 32 + lane) * 8);
                acc[ot] = __builtin_amdgcn_wmma_f32_16x16x32_bf16(
                    false, a, false, b, (short)0, acc[ot], false, false);
            }
        }
        int nn_ = lane & 15;
        int mb  = (lane < 16) ? 0 : 8;
        #pragma unroll
        for (int ot = 0; ot < 4; ++ot)
            #pragma unroll
            for (int r = 0; r < 8; ++r) {
                int m = mb + r, col = ot * 16 + nn_;
                float v = acc[ot][r];
                e_out[(r0 + m) * HD + col] = v + bias[col];
                lds[wave][m * HD + col] = v;
            }
    }
    __syncthreads();
    if (active) {
        int row = lane & 15, head = lane >> 4;
        const float* zr = &lds[wave][row * HD + head * 32];
        float se = 0.f;
        #pragma unroll
        for (int d = 0; d < 32; ++d) se += zr[d] * attn_e[head * 32 + d];
        ee[(r0 + row) * 2 + head] = se;
    }
}

// ---------------------------------------------------------------------------
// Per-(edge,head) logit + segment max via orderable-uint atomicMax.
// ---------------------------------------------------------------------------
__global__ __launch_bounds__(256) void gat_logits_max(
    const int* __restrict__ src, const int* __restrict__ dst,
    const float* __restrict__ el, const float* __restrict__ er,
    const float* __restrict__ ee, float* __restrict__ logit,
    unsigned* __restrict__ mten)
{
    long i = (long)blockIdx.x * 256 + threadIdx.x;
    if (i >= (long)NE * 2) return;
    long e = i >> 1; int h = (int)(i & 1);
    int s = src[e], d = dst[e];
    float v = el[(long)s * 2 + h] + er[(long)d * 2 + h] + ee[i];
    v = v > 0.f ? v : 0.01f * v;
    logit[i] = v;
    atomicMax(&mten[(long)d * 2 + h], enc_f(v));
}

__global__ __launch_bounds__(256) void gat_exp_denom(
    const int* __restrict__ dst, float* __restrict__ logit,
    const unsigned* __restrict__ mten, float* __restrict__ denom)
{
    long i = (long)blockIdx.x * 256 + threadIdx.x;
    if (i >= (long)NE * 2) return;
    long e = i >> 1; int h = (int)(i & 1);
    int d = dst[e];
    float m  = dec_f(mten[(long)d * 2 + h]);
    float ex = __expf(logit[i] - m);
    logit[i] = ex;                       // reuse buffer as ex
    unsafeAtomicAdd(&denom[(long)d * 2 + h], ex);
}

// ---------------------------------------------------------------------------
// SpMM scatter: n_out[dst, j] += alpha * z[src, j]
// ---------------------------------------------------------------------------
__global__ __launch_bounds__(256) void gat_aggregate(
    const int* __restrict__ src, const int* __restrict__ dst,
    const float* __restrict__ ex, const float* __restrict__ denom,
    const float* __restrict__ z, float* __restrict__ n_out)
{
    long i = (long)blockIdx.x * 256 + threadIdx.x;
    if (i >= (long)NE * HD) return;
    long e = i >> 6; int j = (int)(i & 63); int h = j >> 5;
    int s = src[e], d = dst[e];
    float alpha = ex[e * 2 + h] / denom[(long)d * 2 + h];
    float v = alpha * z[(long)s * HD + j];
    unsafeAtomicAdd(&n_out[(long)d * HD + j], v);
}

extern "C" void kernel_launch(void* const* d_in, const int* in_sizes, int n_in,
                              void* d_out, int out_size, void* d_ws, size_t ws_size,
                              hipStream_t stream) {
    const float* nf     = (const float*)d_in[0];
    const float* ef     = (const float*)d_in[1];
    const int*   src    = (const int*)d_in[2];
    const int*   dst    = (const int*)d_in[3];
    const float* W_node = (const float*)d_in[4];
    const float* W_edge = (const float*)d_in[5];
    const float* attn_l = (const float*)d_in[6];
    const float* attn_r = (const float*)d_in[7];
    const float* attn_e = (const float*)d_in[8];
    const float* bias   = (const float*)d_in[9];

    float* n_out = (float*)d_out;
    float* e_out = n_out + (long)NN * HD;

    char* ws = (char*)d_ws;
    unsigned* fragN = (unsigned*)(ws + OFF_FRAG_N);
    unsigned* fragE = (unsigned*)(ws + OFF_FRAG_E);
    float*    z     = (float*)(ws + OFF_Z);
    float*    el    = (float*)(ws + OFF_EL);
    float*    er    = (float*)(ws + OFF_ER);
    float*    ee    = (float*)(ws + OFF_EE);
    float*    logit = (float*)(ws + OFF_LOGIT);
    unsigned* mten  = (unsigned*)(ws + OFF_M);
    float*    denom = (float*)(ws + OFF_DENOM);

    gat_pack_w<<<1, 256, 0, stream>>>(W_node, W_edge, fragN, fragE);

    long init_n = (long)NN * HD;
    gat_init<<<(unsigned)((init_n + 255) / 256), 256, 0, stream>>>(n_out, bias, mten, denom);

    gat_node_gemm<<<(NTILES + 7) / 8, 256, 0, stream>>>(nf, fragN, attn_l, attn_r, z, el, er);
    gat_edge_gemm<<<ETILES / 8, 256, 0, stream>>>(ef, fragE, attn_e, bias, e_out, ee);

    long eh = (long)NE * 2;
    gat_logits_max<<<(unsigned)((eh + 255) / 256), 256, 0, stream>>>(src, dst, el, er, ee, logit, mten);
    gat_exp_denom<<<(unsigned)((eh + 255) / 256), 256, 0, stream>>>(dst, logit, mten, denom);

    long ej = (long)NE * HD;
    gat_aggregate<<<(unsigned)((ej + 255) / 256), 256, 0, stream>>>(src, dst, logit, denom, z, n_out);
}
